// AQLMLinear_15942918602789
// MI455X (gfx1250) — matrix operand order repaired and checked by reading, not compiled, not run on hardware
//
#include <hip/hip_runtime.h>
#include <hip/hip_bf16.h>

// ---------------------------------------------------------------------------
// AQLM linear: Y[8192,4096] = X[8192,4096] @ W^T, W dequantized from 2 additive
// codebooks + row scale. Compute-bound => bf16 WMMA (v_wmma_f32_16x16x32_bf16),
// fp32 accumulation. Staging uses GLOBAL_LOAD_ASYNC_TO_LDS (ASYNCcnt) in a
// 3-buffer rotating pipeline with a single barrier per K-stage.
// ---------------------------------------------------------------------------

typedef __attribute__((ext_vector_type(16))) __bf16 v16bf;
typedef __attribute__((ext_vector_type(8)))  float  v8f;
typedef __attribute__((ext_vector_type(4)))  int    v4i;

union FragB16 { uint4 q[2]; v16bf v; };

#if __has_builtin(__builtin_amdgcn_global_load_async_to_lds_b128)
#define HAVE_ASYNC_LDS 1
#else
#define HAVE_ASYNC_LDS 0
#endif

typedef __attribute__((address_space(1))) v4i* gp128;   // global int4*
typedef __attribute__((address_space(3))) v4i* sp128;   // LDS int4*

// 16B global -> LDS copy. Async DMA path on gfx1250, sync fallback otherwise.
__device__ __forceinline__ void copy16(const unsigned short* g, unsigned short* l) {
#if HAVE_ASYNC_LDS
    __builtin_amdgcn_global_load_async_to_lds_b128((gp128)(void*)g, (sp128)(void*)l, 0, 0);
#else
    *(uint4*)l = *(const uint4*)g;
#endif
}

template <int N>
__device__ __forceinline__ void wait_async() {
#if HAVE_ASYNC_LDS
#if __has_builtin(__builtin_amdgcn_s_wait_asynccnt)
    __builtin_amdgcn_s_wait_asynccnt(N);
#else
    asm volatile("s_wait_asynccnt %0" :: "n"(N) : "memory");
#endif
#endif
}

__device__ __forceinline__ unsigned short f2bf(float f) {
    // round-to-nearest-even fp32 -> bf16
    unsigned int u = __float_as_uint(f);
    unsigned int r = u + 0x7FFFu + ((u >> 16) & 1u);
    return (unsigned short)(r >> 16);
}

// ---------------------------------------------------------------------------
// Kernel 1: X fp32 -> bf16 (4 elements / thread, float4 in, uint2 out)
// ---------------------------------------------------------------------------
__global__ __launch_bounds__(256) void cvt_x_bf16(const float* __restrict__ x,
                                                  unsigned short* __restrict__ xb,
                                                  int n4) {
    int i = blockIdx.x * 256 + threadIdx.x;
    if (i < n4) {
        float4 v = ((const float4*)x)[i];
        uint2 o;
        o.x = (unsigned)f2bf(v.x) | ((unsigned)f2bf(v.y) << 16);
        o.y = (unsigned)f2bf(v.z) | ((unsigned)f2bf(v.w) << 16);
        ((uint2*)xb)[i] = o;
    }
}

// ---------------------------------------------------------------------------
// Kernel 2: dequantize W -> bf16 [4096, 4096].
// One thread per 8-wide group; codebooks (16KB) staged in LDS.
// ---------------------------------------------------------------------------
__global__ __launch_bounds__(256) void dequant_w(const float* __restrict__ codebooks,
                                                 const float* __restrict__ scale,
                                                 const int*   __restrict__ codes,
                                                 unsigned short* __restrict__ wb) {
    __shared__ float cb[2 * 256 * 8];
    for (int i = threadIdx.x; i < 4096; i += 256) cb[i] = codebooks[i];
    __syncthreads();

    int g  = blockIdx.x * 256 + threadIdx.x;  // global group id
    int o  = g >> 9;                          // output row (512 groups / row)
    int gi = g & 511;

    int2 c = *(const int2*)&codes[(size_t)g * 2];
    float s = scale[o];
    const float* p0 = &cb[c.x * 8];
    const float* p1 = &cb[2048 + c.y * 8];

    unsigned int packed[4];
#pragma unroll
    for (int j = 0; j < 4; ++j) {
        float lo = (p0[2 * j]     + p1[2 * j])     * s;
        float hi = (p0[2 * j + 1] + p1[2 * j + 1]) * s;
        packed[j] = (unsigned)f2bf(lo) | ((unsigned)f2bf(hi) << 16);
    }
    *(uint4*)&wb[(size_t)o * 4096 + gi * 8] =
        make_uint4(packed[0], packed[1], packed[2], packed[3]);
}

// ---------------------------------------------------------------------------
// Kernel 3: WMMA GEMM.  C[m,n] = sum_k Xb[m,k] * Wb[n,k]
// Block tile 128x128, BK=32.  8 waves as 2(M) x 4(N); each wave computes a
// 64x32 tile = 4x2 grid of 16x16 wmma accumulators (fp32).
// 3 LDS buffers, async refill issued 2 stages ahead, ONE barrier per stage:
//   top of iter s: wait_asynccnt<=4  (retires stage s; in-order completion)
//                  barrier           (publishes stage s; proves buf (s+2)%3 free)
//                  issue stage s+2 -> buf (s+2)%3 ; compute stage s from buf s%3
// Row stride 40 bf16 (pad 8) keeps fragment b128 ds_loads bank-conflict-free.
// ---------------------------------------------------------------------------
#define BM 128
#define BN 128
#define BK 32
#define LDSS 40
#define BUFE (BM * LDSS)     // elements per buffer per operand
#define NBUF 3
#define STAGES (4096 / BK)   // 128

__global__ __launch_bounds__(256) void gemm_bf16_wmma(const unsigned short* __restrict__ Xb,
                                                      const unsigned short* __restrict__ Wb,
                                                      float* __restrict__ out) {
    __shared__ __align__(16) unsigned short Xs[NBUF][BUFE];   // 30 KB
    __shared__ __align__(16) unsigned short Ws[NBUF][BUFE];   // 30 KB

    const int tid   = threadIdx.x;
    const int lane  = tid & 31;
    const int wave  = tid >> 5;
    const int waveM = wave >> 2;            // 0..1 -> 64-row slab
    const int waveN = wave & 3;             // 0..3 -> 32-col slab
    const int m0 = blockIdx.y * BM;
    const int n0 = blockIdx.x * BN;

    // ---- staging copy mapping: 128 rows x 4 x 16B chunks; 2 chunks/thread/op
    const int cr = tid >> 2;                // 0..63
    const int cc = (tid & 3) * 8;           // element col 0,8,16,24
    const unsigned short* gx0 = Xb + (size_t)(m0 + cr) * 4096 + cc;
    const unsigned short* gx1 = gx0 + (size_t)64 * 4096;
    const unsigned short* gw0 = Wb + (size_t)(n0 + cr) * 4096 + cc;
    const unsigned short* gw1 = gw0 + (size_t)64 * 4096;
    unsigned short* lx0 = &Xs[0][cr * LDSS + cc];
    unsigned short* lw0 = &Ws[0][cr * LDSS + cc];

    // ---- fragment addressing (per ISA WMMA VGPR layout tables)
    const int rowA  = lane & 15;            // A: M = lane%16
    const int halfK = (lane >> 4) * 8;      // A: lanes16-31 hold K+8 halves
    const int colB  = lane & 15;            // B: N = lane%16
    const int kselB = (lane >> 4) * 8;      // B: lanes16-31 hold K+8 halves
    const int pA = (waveM * 64 + rowA) * LDSS;
    const int pB = (waveN * 32 + colB) * LDSS + kselB;

    v8f acc[4][2];
#pragma unroll
    for (int i = 0; i < 4; ++i)
#pragma unroll
        for (int j = 0; j < 2; ++j)
            acc[i][j] = (v8f){0.f, 0.f, 0.f, 0.f, 0.f, 0.f, 0.f, 0.f};

    // issue one stage's 4 async b128 copies into buffer `buf`
    auto issue = [&](int s, int buf) {
        const int k   = s * BK;
        const int off = buf * BUFE;
        copy16(gx0 + k, lx0 + off);
        copy16(gx1 + k, lx0 + off + 64 * LDSS);
        copy16(gw0 + k, lw0 + off);
        copy16(gw1 + k, lw0 + off + 64 * LDSS);
    };

    issue(0, 0);
    issue(1, 1);

    int buf  = 0;   // s % 3
    int ibuf = 2;   // (s+2) % 3
    for (int s = 0; s < STAGES; ++s) {
        // retire stage s's 4 copies (the <=4 remaining are stage s+1's)
        if (s + 1 < STAGES) wait_async<4>();
        else                wait_async<0>();
        __syncthreads();    // publish stage s; all waves done reading buf ibuf

        if (s + 2 < STAGES) issue(s + 2, ibuf);

        const unsigned short* xs = &Xs[buf][0];
        const unsigned short* ws = &Ws[buf][0];

        FragB16 a[4], b[2];
#pragma unroll
        for (int j = 0; j < 2; ++j) {
            const unsigned short* p = ws + pB + j * 16 * LDSS;
            b[j].q[0] = *(const uint4*)p;
            b[j].q[1] = *(const uint4*)(p + 16);
        }
#pragma unroll
        for (int i = 0; i < 4; ++i) {
            const unsigned short* p = xs + pA + i * 16 * LDSS;
            a[i].q[0] = *(const uint4*)(p + halfK);
            a[i].q[1] = *(const uint4*)(p + 16 + halfK);
        }
#pragma unroll
        for (int i = 0; i < 4; ++i)
#pragma unroll
            for (int j = 0; j < 2; ++j)
                acc[i][j] = __builtin_amdgcn_wmma_f32_16x16x32_bf16(
                    false, a[i].v, false, b[j].v, (short)0, acc[i][j], false, false);

        // rotate buffer indices (avoid % in the hot loop)
        buf  = (buf  == NBUF - 1) ? 0 : buf  + 1;
        ibuf = (ibuf == NBUF - 1) ? 0 : ibuf + 1;
    }

    // write back: acc VGPR r -> (M = r + 8*(lane/16), N = lane%16)
    const int mOff = m0 + waveM * 64 + ((lane >> 4) << 3);
    const int nOff = n0 + waveN * 32 + (lane & 15);
#pragma unroll
    for (int i = 0; i < 4; ++i)
#pragma unroll
        for (int j = 0; j < 2; ++j)
#pragma unroll
            for (int r = 0; r < 8; ++r)
                out[(size_t)(mOff + i * 16 + r) * 4096 + nOff + j * 16] = acc[i][j][r];
}

// ---------------------------------------------------------------------------
// launch
// ---------------------------------------------------------------------------
extern "C" void kernel_launch(void* const* d_in, const int* in_sizes, int n_in,
                              void* d_out, int out_size, void* d_ws, size_t ws_size,
                              hipStream_t stream) {
    const float* x         = (const float*)d_in[0];   // [4,2048,4096]
    const float* codebooks = (const float*)d_in[1];   // [2,256,8]
    const float* scale     = (const float*)d_in[2];   // [4096,1]
    const int*   codes     = (const int*)d_in[3];     // [4096,512,2]
    float* out = (float*)d_out;                       // [4,2048,4096]

    unsigned short* Xbf = (unsigned short*)d_ws;                 // 64MB bf16
    unsigned short* Wbf = Xbf + (size_t)8192 * 4096;             // 32MB bf16

    cvt_x_bf16<<<(8192 * 4096 / 4 + 255) / 256, 256, 0, stream>>>(x, Xbf, 8192 * 4096 / 4);
    dequant_w<<<(4096 * 512) / 256, 256, 0, stream>>>(codebooks, scale, codes, Wbf);

    dim3 grid(4096 / BN, 8192 / BM);   // (32, 64)
    gemm_bf16_wmma<<<grid, 256, 0, stream>>>(Xbf, Wbf, out);
}
